// MS_MSA_75333726372345
// MI455X (gfx1250) — matrix operand order repaired and checked by reading, not compiled, not run on hardware
//
#include <hip/hip_runtime.h>
#include <hip/hip_bf16.h>

#define BATCH 4
#define HH 256
#define WWD 256
#define CDIM 64
#define NPIX (HH * WWD)          // 65536
#define TOTPIX (BATCH * NPIX)    // 262144

#if defined(__HIP_DEVICE_COMPILE__) && defined(__gfx1250__) && \
    __has_builtin(__builtin_amdgcn_global_load_async_to_lds_b128) && \
    __has_builtin(__builtin_amdgcn_s_wait_asynccnt)
#define USE_ASYNC_LDS 1
#endif

typedef _Float16 v16h __attribute__((ext_vector_type(16)));
typedef float    v8f  __attribute__((ext_vector_type(8)));

#ifdef USE_ASYNC_LDS
// exact parameter type from the compiler diagnostic: vector of 4 ints
typedef int vsi4 __attribute__((__vector_size__(4 * sizeof(int))));
typedef __attribute__((address_space(1))) vsi4* as1_v4i_p;
typedef __attribute__((address_space(3))) vsi4* as3_v4i_p;
#endif

union Af { v16h v; _Float16 e[16]; };
union Bf { v16h v; unsigned  u[8]; };
union Cf { v8f  v; float     e[8]; };
union H16 { _Float16 f; unsigned short u; };

static __device__ __forceinline__ unsigned pack2(float a, float b) {
    H16 h0, h1; h0.f = (_Float16)a; h1.f = (_Float16)b;
    return (unsigned)h0.u | ((unsigned)h1.u << 16);
}

// ---------------------------------------------------------------------------
// K0: pack weights into WMMA f16 B-operand layout.
// B 16-bit 32x16 (KxN): lanes 0-15 hold N=lane, K = (v<4?0:16)+(v&3)*2+h; lanes 16-31 hold
// N=lane-16, K = 8+(v<4?0:16)+(v&3)*2+h (plus ktile*32).
// packQKV: [nt(5)][kt(2)][lane(32)][v(8)]  N: 0..63 = Wv rows, 64..71 = Wq[0:8], 72..79 = Wk[0:8]
// packPP : [nt(4)][kt(2)][lane(32)][v(8)]  N = Wpp row (output channel)
// ---------------------------------------------------------------------------
__global__ void k_prep(const float* __restrict__ Wq, const float* __restrict__ bq,
                       const float* __restrict__ Wk, const float* __restrict__ bk,
                       const float* __restrict__ Wv, const float* __restrict__ bv,
                       const float* __restrict__ Wpp,
                       unsigned* __restrict__ packQKV, unsigned* __restrict__ packPP,
                       float* __restrict__ biasC)
{
    const int tid = threadIdx.x;
    for (int idx = tid; idx < 5 * 2 * 32 * 8; idx += blockDim.x) {
        int v = idx & 7, lane = (idx >> 3) & 31, kt = (idx >> 8) & 1, nt = idx >> 9;
        int N = nt * 16 + (lane & 15);
        int Kb = kt * 32 + ((v < 4) ? 0 : 16) + (v & 3) * 2 + ((lane >= 16) ? 8 : 0);
        float w0, w1;
        if (N < 64)      { w0 = Wv[N * 64 + Kb];        w1 = Wv[N * 64 + Kb + 1]; }
        else if (N < 72) { w0 = Wq[(N - 64) * 64 + Kb]; w1 = Wq[(N - 64) * 64 + Kb + 1]; }
        else             { w0 = Wk[(N - 72) * 64 + Kb]; w1 = Wk[(N - 72) * 64 + Kb + 1]; }
        packQKV[idx] = pack2(w0, w1);
    }
    for (int idx = tid; idx < 4 * 2 * 32 * 8; idx += blockDim.x) {
        int v = idx & 7, lane = (idx >> 3) & 31, kt = (idx >> 8) & 1, nt = idx >> 9;
        int N = nt * 16 + (lane & 15);
        int Kb = kt * 32 + ((v < 4) ? 0 : 16) + (v & 3) * 2 + ((lane >= 16) ? 8 : 0);
        packPP[idx] = pack2(Wpp[N * 64 + Kb], Wpp[N * 64 + Kb + 1]);
    }
    for (int i = tid; i < 80; i += blockDim.x)
        biasC[i] = (i < 64) ? bv[i] : (i < 72 ? bq[i - 64] : bk[i - 72]);
}

// ---------------------------------------------------------------------------
// K1: fused QKV projection GEMM.  D(16pix x 80ch) = X(16x64) * W^T, bias in C.
// A 16-bit 16x32 layout: lane l (m=l&15, g=l>>4), vgpr v, half h:
//   K = kc*32 + (v<4?0:16) + (v&3)*2 + h + g*8
// ---------------------------------------------------------------------------
__global__ void k_qkv(const float* __restrict__ x, const unsigned* __restrict__ packQKV,
                      const float* __restrict__ biasC,
                      float* __restrict__ v_inp, float* __restrict__ q8,
                      float* __restrict__ k8)
{
    const int wave = threadIdx.x >> 5;
    const int lane = threadIdx.x & 31;
    const long tile = (long)blockIdx.x * (blockDim.x >> 5) + wave;   // 0..16383
    const long base = tile * 16;
    const int g = lane >> 4, m = lane & 15;
    const float* xr = x + (base + m) * 64;
    __builtin_prefetch(xr + 128 * 64, 0, 3);   // global_prefetch_b8: ~8 tiles ahead

    Af af[2];
#pragma unroll
    for (int kc = 0; kc < 2; ++kc)
#pragma unroll
        for (int v = 0; v < 8; ++v) {
            int Kb = kc * 32 + ((v < 4) ? 0 : 16) + (v & 3) * 2 + g * 8;
            float2 f2 = *(const float2*)(xr + Kb);
            af[kc].e[2 * v]     = (_Float16)f2.x;
            af[kc].e[2 * v + 1] = (_Float16)f2.y;
        }

#pragma unroll
    for (int nt = 0; nt < 5; ++nt) {
        Cf cf;
        float bb = biasC[nt * 16 + m];
#pragma unroll
        for (int r = 0; r < 8; ++r) cf.e[r] = bb;
#pragma unroll
        for (int kc = 0; kc < 2; ++kc) {
            Bf bf;
            const uint4* pb = (const uint4*)(packQKV + (((nt * 2 + kc) * 32 + lane) << 3));
            uint4 u0 = pb[0], u1 = pb[1];
            bf.u[0] = u0.x; bf.u[1] = u0.y; bf.u[2] = u0.z; bf.u[3] = u0.w;
            bf.u[4] = u1.x; bf.u[5] = u1.y; bf.u[6] = u1.z; bf.u[7] = u1.w;
            cf.v = __builtin_amdgcn_wmma_f32_16x16x32_f16(
                false, af[kc].v, false, bf.v, (short)0, cf.v, false, false);
        }
        if (nt < 4) {
#pragma unroll
            for (int r = 0; r < 8; ++r) {
                long p = base + r + 8 * g;                 // D: M = r + 8*(lane>>4)
                v_inp[p * 64 + nt * 16 + m] = cf.e[r];
            }
        } else {
#pragma unroll
            for (int r = 0; r < 8; ++r) {
                long p = base + r + 8 * g;
                if (m < 8) q8[p * 8 + m] = cf.e[r];
                else       k8[p * 8 + (m - 8)] = cf.e[r];
            }
        }
    }
}

// ---------------------------------------------------------------------------
// K2: partial Gram (8x8) + squared norms over 1024-pixel chunks (deterministic
// two-stage reduction in pure f32; K=65536 so precision matters here).
// ---------------------------------------------------------------------------
__global__ void k_gram(const float* __restrict__ q8, const float* __restrict__ k8,
                       float* __restrict__ partial)
{
    const int b = blockIdx.y, chunk = blockIdx.x, tid = threadIdx.x;
    const long base = (long)b * NPIX + (long)chunk * 1024;
    float G[64], qs[8], ks[8];
#pragma unroll
    for (int i = 0; i < 64; ++i) G[i] = 0.f;
#pragma unroll
    for (int i = 0; i < 8; ++i) { qs[i] = 0.f; ks[i] = 0.f; }

    for (int i = tid; i < 1024; i += 256) {
        const float4* qp = (const float4*)(q8 + (base + i) * 8);
        const float4* kp = (const float4*)(k8 + (base + i) * 8);
        float4 qa = qp[0], qb = qp[1], ka = kp[0], kb = kp[1];
        float qv[8] = {qa.x, qa.y, qa.z, qa.w, qb.x, qb.y, qb.z, qb.w};
        float kv[8] = {ka.x, ka.y, ka.z, ka.w, kb.x, kb.y, kb.z, kb.w};
#pragma unroll
        for (int c = 0; c < 8; ++c)
#pragma unroll
            for (int d = 0; d < 8; ++d) G[c * 8 + d] += kv[c] * qv[d];
#pragma unroll
        for (int d = 0; d < 8; ++d) { qs[d] += qv[d] * qv[d]; ks[d] += kv[d] * kv[d]; }
    }

    __shared__ float red[256];
    for (int i = 0; i < 80; ++i) {
        red[tid] = (i < 64) ? G[i] : (i < 72 ? qs[i - 64] : ks[i - 72]);
        __syncthreads();
        for (int s = 128; s > 0; s >>= 1) {
            if (tid < s) red[tid] += red[tid + s];
            __syncthreads();
        }
        if (tid == 0) partial[((long)(b * 64 + chunk)) * 80 + i] = red[0];
        __syncthreads();
    }
}

// ---------------------------------------------------------------------------
// K3: final reduce, normalize, per-head softmax, fold through Wp -> E[b][8][64]
// ---------------------------------------------------------------------------
__global__ void k_finalize(const float* __restrict__ partial,
                           const float* __restrict__ rescale,
                           const float* __restrict__ Wp, float* __restrict__ E)
{
    __shared__ float g[80];
    __shared__ float attnS[64][8];
    const int tid = threadIdx.x;   // 128 threads
    for (int b = 0; b < BATCH; ++b) {
        if (tid < 80) {
            float s = 0.f;
            for (int blk = 0; blk < 64; ++blk)
                s += partial[((long)(b * 64 + blk)) * 80 + tid];
            g[tid] = s;
        }
        __syncthreads();
        if (tid < 64) {
            int j = tid >> 3, c = tid & 7;
            float scale = rescale[j] * 0.35355339059327373f;   // 1/sqrt(dh)
            float kn = fmaxf(sqrtf(g[72 + c]), 1e-12f);
            float a[8], mx = -1e30f;
#pragma unroll
            for (int d = 0; d < 8; ++d) {
                float qn = fmaxf(sqrtf(g[64 + d]), 1e-12f);
                a[d] = g[c * 8 + d] / (kn * qn) * scale;
                mx = fmaxf(mx, a[d]);
            }
            float sum = 0.f;
#pragma unroll
            for (int d = 0; d < 8; ++d) { a[d] = __expf(a[d] - mx); sum += a[d]; }
            float inv = 1.f / sum;
#pragma unroll
            for (int d = 0; d < 8; ++d) attnS[tid][d] = a[d] * inv;
        }
        __syncthreads();
        if (tid < 64) {   // tid = output channel o;  m-index == j*8+c == t2
#pragma unroll
            for (int d = 0; d < 8; ++d) {
                float s = 0.f;
                for (int t2 = 0; t2 < 64; ++t2) s += attnS[t2][d] * Wp[tid * 64 + t2];
                E[b * 512 + d * 64 + tid] = s;
            }
        }
        __syncthreads();
    }
}

// ---------------------------------------------------------------------------
// K4: depthwise conv1 (dil 1 for c<32, dil 3 for c>=32) + exact GELU, NHWC.
// One thread = one pixel x 4 channels (float4).
// ---------------------------------------------------------------------------
__global__ void k_conv1(const float* __restrict__ vin, const float* __restrict__ w1a,
                        const float* __restrict__ w1b, float* __restrict__ t)
{
    const long tid = (long)blockIdx.x * blockDim.x + threadIdx.x;  // TOTPIX*16
    const int cq = (int)(tid & 15);
    const long pix = tid >> 4;
    const int b = (int)(pix >> 16);
    const int rem = (int)(pix & 65535);
    const int y = rem >> 8, x = rem & 255;
    const int c0 = cq * 4;
    const int dil = (c0 < 32) ? 1 : 3;
    const float* wsrc = (c0 < 32) ? (w1a + c0 * 9) : (w1b + (c0 - 32) * 9);
    float w[4][9];
#pragma unroll
    for (int cc = 0; cc < 4; ++cc)
#pragma unroll
        for (int k = 0; k < 9; ++k) w[cc][k] = wsrc[cc * 9 + k];

    float ax = 0.f, ay = 0.f, az = 0.f, aw = 0.f;
#pragma unroll
    for (int dy = -1; dy <= 1; ++dy) {
        int yy = y + dy * dil;
        if (yy < 0 || yy >= HH) continue;
#pragma unroll
        for (int dx = -1; dx <= 1; ++dx) {
            int xx = x + dx * dil;
            if (xx < 0 || xx >= WWD) continue;
            float4 vv = *(const float4*)(vin + (((long)b * NPIX + yy * WWD + xx) << 6) + c0);
            int k = (dy + 1) * 3 + dx + 1;
            ax += vv.x * w[0][k]; ay += vv.y * w[1][k];
            az += vv.z * w[2][k]; aw += vv.w * w[3][k];
        }
    }
    float4 o;
    o.x = 0.5f * ax * (1.f + erff(ax * 0.70710678118654752f));
    o.y = 0.5f * ay * (1.f + erff(ay * 0.70710678118654752f));
    o.z = 0.5f * az * (1.f + erff(az * 0.70710678118654752f));
    o.w = 0.5f * aw * (1.f + erff(aw * 0.70710678118654752f));
    *(float4*)(t + (pix << 6) + c0) = o;
}

// ---------------------------------------------------------------------------
// K5: fused conv2(+bias) -> f16 A-operand -> WMMA x Wpp^T, C-init = out_c+bp+bpp.
// Block: 512 threads = 16 waves, each wave = one row of a 16x16 pixel tile.
// LDS halo tile: 22 x 22 x 64 f32 (123,904 B of the 320 KB/WGP LDS).
// Halo staging uses GLOBAL_LOAD_ASYNC_TO_LDS_B128 (ASYNCcnt) when available.
// ---------------------------------------------------------------------------
__global__ void __launch_bounds__(512)
k_fuse(const float* __restrict__ t, const float* __restrict__ vin,
       const float* __restrict__ w2a, const float* __restrict__ w2b,
       const float* __restrict__ b2a, const float* __restrict__ b2b,
       const float* __restrict__ E, const float* __restrict__ bp,
       const float* __restrict__ bpp, const unsigned* __restrict__ packPP,
       float* __restrict__ out)
{
    extern __shared__ float lds[];     // 22*22*64 floats
    const int b = blockIdx.z, tX = blockIdx.x, tY = blockIdx.y;
    const int tid = threadIdx.x;
    const long pbase = (long)b * NPIX;

    for (int idx = tid; idx < 22 * 22 * 16; idx += 512) {
        int c4 = idx & 15;
        int col = (idx >> 4) % 22;
        int row = (idx >> 4) / 22;
        int gy = tY * 16 + row - 3, gx = tX * 16 + col - 3;
        float* lp = lds + ((row * 22 + col) << 6) + c4 * 4;
        bool inb = (gy >= 0 && gy < HH && gx >= 0 && gx < WWD);
#ifdef USE_ASYNC_LDS
        if (inb) {
            const float* gp = t + ((pbase + gy * WWD + gx) << 6) + c4 * 4;
            __builtin_amdgcn_global_load_async_to_lds_b128(
                (as1_v4i_p)gp, (as3_v4i_p)lp, 0, 0);
        } else {
            float4 z = {0.f, 0.f, 0.f, 0.f};
            *(float4*)lp = z;
        }
#else
        float4 vv = {0.f, 0.f, 0.f, 0.f};
        if (inb) vv = *(const float4*)(t + ((pbase + gy * WWD + gx) << 6) + c4 * 4);
        *(float4*)lp = vv;
#endif
    }
#ifdef USE_ASYNC_LDS
    __builtin_amdgcn_s_wait_asynccnt(0);
#endif
    __syncthreads();

    const int wave = tid >> 5;          // row within tile
    const int lane = tid & 31;
    const int g = lane >> 4, m = lane & 15;
    const int lrow = 3 + wave, lcol = 3 + m;

    Af af[2];
#pragma unroll
    for (int kc = 0; kc < 2; ++kc)
#pragma unroll
        for (int v = 0; v < 8; ++v)
#pragma unroll
            for (int h = 0; h < 2; ++h) {
                int ch = kc * 32 + ((v < 4) ? 0 : 16) + (v & 3) * 2 + h + g * 8;
                int dil; const float* wsrc; float s;
                if (ch < 32) { dil = 1; wsrc = w2a + ch * 9;        s = b2a[ch]; }
                else         { dil = 3; wsrc = w2b + (ch - 32) * 9; s = b2b[ch - 32]; }
#pragma unroll
                for (int dy = -1; dy <= 1; ++dy)
#pragma unroll
                    for (int dx = -1; dx <= 1; ++dx)
                        s += lds[(((lrow + dy * dil) * 22 + (lcol + dx * dil)) << 6) + ch]
                             * wsrc[(dy + 1) * 3 + dx + 1];
                af[kc].e[2 * v + h] = (_Float16)s;
            }

    const int gy = tY * 16 + wave;
#pragma unroll
    for (int nt = 0; nt < 4; ++nt) {
        const int o = nt * 16 + m;
        float eo[8];
#pragma unroll
        for (int d = 0; d < 8; ++d) eo[d] = E[b * 512 + d * 64 + o];
        const float bb = bp[o] + bpp[o];
        Cf cf;
#pragma unroll
        for (int r = 0; r < 8; ++r) {
            long gp = pbase + gy * WWD + (tX * 16 + r + 8 * g);
            const float* v8p = vin + (gp << 6);
            float s = bb;
#pragma unroll
            for (int d = 0; d < 8; ++d) s += v8p[d] * eo[d];
            cf.e[r] = s;
        }
#pragma unroll
        for (int kc = 0; kc < 2; ++kc) {
            Bf bf;
            const uint4* pb = (const uint4*)(packPP + (((nt * 2 + kc) * 32 + lane) << 3));
            uint4 u0 = pb[0], u1 = pb[1];
            bf.u[0] = u0.x; bf.u[1] = u0.y; bf.u[2] = u0.z; bf.u[3] = u0.w;
            bf.u[4] = u1.x; bf.u[5] = u1.y; bf.u[6] = u1.z; bf.u[7] = u1.w;
            cf.v = __builtin_amdgcn_wmma_f32_16x16x32_f16(
                false, af[kc].v, false, bf.v, (short)0, cf.v, false, false);
        }
#pragma unroll
        for (int r = 0; r < 8; ++r) {
            long gp = pbase + gy * WWD + (tX * 16 + r + 8 * g);
            out[(gp << 6) + o] = cf.e[r];
        }
    }
}

// ---------------------------------------------------------------------------
extern "C" void kernel_launch(void* const* d_in, const int* in_sizes, int n_in,
                              void* d_out, int out_size, void* d_ws, size_t ws_size,
                              hipStream_t stream)
{
    const float* x      = (const float*)d_in[0];
    const float* Wq     = (const float*)d_in[1];
    const float* bq     = (const float*)d_in[2];
    const float* Wk     = (const float*)d_in[3];
    const float* bk     = (const float*)d_in[4];
    const float* Wv     = (const float*)d_in[5];
    const float* bv     = (const float*)d_in[6];
    const float* rsc    = (const float*)d_in[7];
    const float* Wp     = (const float*)d_in[8];
    const float* bp     = (const float*)d_in[9];
    const float* Wpp    = (const float*)d_in[10];
    const float* bpp    = (const float*)d_in[11];
    const float* pe1_w1 = (const float*)d_in[12];
    const float* pe1_w2 = (const float*)d_in[13];
    const float* pe1_b2 = (const float*)d_in[14];
    const float* pe2_w1 = (const float*)d_in[15];
    const float* pe2_w2 = (const float*)d_in[16];
    const float* pe2_b2 = (const float*)d_in[17];
    float* out = (float*)d_out;

    float* ws      = (float*)d_ws;
    float* v_inp   = ws;                          // TOTPIX*64
    float* q8      = v_inp + (long)TOTPIX * 64;   // TOTPIX*8
    float* k8      = q8 + (long)TOTPIX * 8;       // TOTPIX*8
    float* tbuf    = k8 + (long)TOTPIX * 8;       // TOTPIX*64
    float* partial = tbuf + (long)TOTPIX * 64;    // 4*64*80
    float* Ebuf    = partial + 4 * 64 * 80;       // 4*8*64
    float* biasC   = Ebuf + 4 * 512;              // 80
    unsigned* packQKV = (unsigned*)(biasC + 80);  // 2560
    unsigned* packPP  = packQKV + 2560;           // 2048

    k_prep<<<1, 256, 0, stream>>>(Wq, bq, Wk, bk, Wv, bv, Wpp, packQKV, packPP, biasC);
    k_qkv<<<2048, 256, 0, stream>>>(x, packQKV, biasC, v_inp, q8, k8);
    k_gram<<<dim3(64, BATCH), 256, 0, stream>>>(q8, k8, partial);
    k_finalize<<<1, 128, 0, stream>>>(partial, rsc, Wp, Ebuf);
    k_conv1<<<(TOTPIX * 16) / 256, 256, 0, stream>>>(v_inp, pe1_w1, pe2_w1, tbuf);
    k_fuse<<<dim3(WWD / 16, HH / 16, BATCH), 512, 22 * 22 * 64 * sizeof(float), stream>>>(
        tbuf, v_inp, pe1_w2, pe2_w2, pe1_b2, pe2_b2, Ebuf, bp, bpp, packPP, out);
}
